// RNN_alpha_89928025244450
// MI455X (gfx1250) — matrix operand order repaired and checked by reading, not compiled, not run on hardware
//
#include <hip/hip_runtime.h>
#include <hip/hip_bf16.h>

// Problem dims (fixed by reference setup_inputs)
#define BB 4
#define SS 2048
#define HH 1024
#define LL 4
#define FF 4096
#define NCK 32         // scan chunks
#define CHL 64         // SS / NCK

typedef __attribute__((ext_vector_type(16))) __bf16 bf16x16;
typedef __attribute__((ext_vector_type(8)))  __bf16 bf16x8;
typedef __attribute__((ext_vector_type(8)))  float  f32x8;

union BF16Frag { bf16x16 v; bf16x8 h2[2]; };

// ---- CDNA5 async global->LDS staging (ASYNCcnt-tracked), with sync fallback
#if defined(__HIP_DEVICE_COMPILE__) && __has_builtin(__builtin_amdgcn_global_load_async_to_lds_b128)
#define USE_ASYNC 1
#else
#define USE_ASYNC 0
#endif

#if USE_ASYNC
typedef int v4i __attribute__((vector_size(16)));
typedef __attribute__((address_space(1))) v4i as1_v4i;   // global
typedef __attribute__((address_space(3))) v4i as3_v4i;   // LDS
#endif

__device__ __forceinline__ void lds_async16(const __bf16* g, __bf16* l) {
#if USE_ASYNC
  __builtin_amdgcn_global_load_async_to_lds_b128((as1_v4i*)g, (as3_v4i*)l, 0, 0);
#else
  *(bf16x8*)l = *(const bf16x8*)g;
#endif
}

__device__ __forceinline__ void async_wait0() {
#if USE_ASYNC
#if __has_builtin(__builtin_amdgcn_s_wait_asynccnt)
  __builtin_amdgcn_s_wait_asynccnt(0);
#else
  asm volatile("s_wait_asynccnt 0x0" ::: "memory");
#endif
#endif
}

// ---------------------------------------------------------------- utilities
__global__ void cvt_f32_bf16_k(const float* __restrict__ x, __bf16* __restrict__ y,
                               size_t n, float mul) {
  size_t i = (size_t)blockIdx.x * blockDim.x + threadIdx.x;
  size_t stride = (size_t)gridDim.x * blockDim.x;
  for (; i < n; i += stride) y[i] = (__bf16)(x[i] * mul);
}

__global__ void gamma_k(const float* __restrict__ nu, float* __restrict__ g, int n) {
  int i = blockIdx.x * blockDim.x + threadIdx.x;
  if (i < n) {
    float mag2 = expf(-2.0f * expf(nu[i]));            // |lam|^2
    g[i] = sqrtf(fmaxf(1.0f - mag2, 0.0f));
  }
}

__global__ void embed_k(const int* __restrict__ ids, const float* __restrict__ we,
                        const float* __restrict__ pe, float* __restrict__ h,
                        __bf16* __restrict__ hbf) {
  size_t i = (size_t)blockIdx.x * blockDim.x + threadIdx.x;
  size_t total = (size_t)BB * SS * HH;
  if (i >= total) return;
  int hh = (int)(i % HH);
  size_t bs = i / HH;
  int s = (int)(bs % SS);
  int tok = ids[bs];
  float v = we[(size_t)tok * HH + hh] + pe[(size_t)s * HH + hh];
  h[i] = v;
  hbf[i] = (__bf16)v;
}

// ---------------------------------------------------------------- bf16 WMMA GEMM
// C[M,N] = A[M,K](bf16,row-major) x W[N,K](bf16,row-major => B^T along K)
// MODE 0: out_f32 = acc * aux[n]          (gamma column scale -> scan buffers)
// MODE 1: out_bf16 = gelu(acc + aux[n])   (FFN up-proj)
// MODE 2: out_f32 = acc + aux[n]          (FFN down-proj)
// Double-buffered LDS; async global->LDS loads of tile k+1 overlap WMMAs of tile k.
template <int MODE>
__global__ __launch_bounds__(256) void gemm_bf16_k(
    const __bf16* __restrict__ A, const __bf16* __restrict__ W,
    const float* __restrict__ aux, void* __restrict__ outp, int N, int K) {
  __shared__ __bf16 As[2][128][40];   // 128x32 tile, +8 pad per row
  __shared__ __bf16 Ws[2][128][40];
  const int tid = threadIdx.x;
  const int wave = tid >> 5;
  const int lane = tid & 31;
  const int hl = lane >> 4;        // 0/1: half-wave select
  const int l15 = lane & 15;
  const int bm = blockIdx.x * 128;
  const int bn = blockIdx.y * 128;

  const int r0 = tid >> 2;            // staging: rows r0 and r0+64, 8 bf16 chunks
  const int c0 = (tid & 3) * 8;

  f32x8 acc[8];
  const f32x8 zero = {0.f, 0.f, 0.f, 0.f, 0.f, 0.f, 0.f, 0.f};
#pragma unroll
  for (int j = 0; j < 8; ++j) acc[j] = zero;

  const int aRow = wave * 16 + l15;  // this wave's 16 output rows
  const int aCol = hl * 8;           // K sub-block per ISA 16-bit A layout
  const int wCol = hl * 8;           // K sub-block per B layout

  auto stage = [&](int buf, int k0) {
    lds_async16(A + (size_t)(bm + r0) * K + k0 + c0,      &As[buf][r0][c0]);
    lds_async16(A + (size_t)(bm + r0 + 64) * K + k0 + c0, &As[buf][r0 + 64][c0]);
    lds_async16(W + (size_t)(bn + r0) * K + k0 + c0,      &Ws[buf][r0][c0]);
    lds_async16(W + (size_t)(bn + r0 + 64) * K + k0 + c0, &Ws[buf][r0 + 64][c0]);
  };

  stage(0, 0);
  async_wait0();
  __syncthreads();

  const int nk = K >> 5;
  for (int kk = 0; kk < nk; ++kk) {
    const int buf = kk & 1;
    if (kk + 1 < nk) stage(buf ^ 1, (kk + 1) << 5);   // prefetch next tile (async)
    BF16Frag af;                                       // A frag: contiguous K chunks
    af.h2[0] = *(const bf16x8*)&As[buf][aRow][aCol];
    af.h2[1] = *(const bf16x8*)&As[buf][aRow][aCol + 16];
#pragma unroll
    for (int j = 0; j < 8; ++j) {                      // 8 column tiles, A reused
      BF16Frag wf;
      wf.h2[0] = *(const bf16x8*)&Ws[buf][j * 16 + l15][wCol];
      wf.h2[1] = *(const bf16x8*)&Ws[buf][j * 16 + l15][wCol + 16];
      acc[j] = __builtin_amdgcn_wmma_f32_16x16x32_bf16(
          false, af.v, false, wf.v, (short)0, acc[j], false, false);
    }
    if (kk + 1 < nk) {
      async_wait0();      // next tile fully in LDS
      __syncthreads();    // all waves done reading current buffer
    }
  }

  const int mBase = bm + wave * 16 + hl * 8;  // C/D layout: vgpr r -> row r (+8 hi half)
#pragma unroll
  for (int j = 0; j < 8; ++j) {
    const int n = bn + j * 16 + l15;
    const float an = aux[n];
#pragma unroll
    for (int r = 0; r < 8; ++r) {
      float v = acc[j][r];
      size_t o = (size_t)(mBase + r) * N + n;
      if (MODE == 0) {
        ((float*)outp)[o] = v * an;
      } else if (MODE == 1) {
        v += an;
        float g = 0.5f * v * (1.0f + erff(v * 0.70710678118f));
        ((__bf16*)outp)[o] = (__bf16)g;
      } else {
        ((float*)outp)[o] = v + an;
      }
    }
  }
}

// ---------------------------------------------------------------- complex projection GEMM
// out[M,N] = Sre[M,K] x Wre[N,K] + Sim[M,K] x WimNeg[N,K]   (WimNeg = -C_im, all bf16)
__global__ __launch_bounds__(256) void gemm_pair_k(
    const __bf16* __restrict__ Are, const __bf16* __restrict__ Aim,
    const __bf16* __restrict__ Wre, const __bf16* __restrict__ WimN,
    float* __restrict__ outp, int N, int K) {
  __shared__ __bf16 Ar[2][128][40];
  __shared__ __bf16 Ai[2][128][40];
  __shared__ __bf16 Wr[2][128][40];
  __shared__ __bf16 Wi[2][128][40];
  const int tid = threadIdx.x;
  const int wave = tid >> 5;
  const int lane = tid & 31;
  const int hl = lane >> 4;
  const int l15 = lane & 15;
  const int bm = blockIdx.x * 128;
  const int bn = blockIdx.y * 128;

  const int r0 = tid >> 2;
  const int c0 = (tid & 3) * 8;

  f32x8 acc[8];
  const f32x8 zero = {0.f, 0.f, 0.f, 0.f, 0.f, 0.f, 0.f, 0.f};
#pragma unroll
  for (int j = 0; j < 8; ++j) acc[j] = zero;

  const int aRow = wave * 16 + l15;
  const int aCol = hl * 8;
  const int wCol = hl * 8;

  auto stage = [&](int buf, int k0) {
    lds_async16(Are + (size_t)(bm + r0) * K + k0 + c0,       &Ar[buf][r0][c0]);
    lds_async16(Are + (size_t)(bm + r0 + 64) * K + k0 + c0,  &Ar[buf][r0 + 64][c0]);
    lds_async16(Aim + (size_t)(bm + r0) * K + k0 + c0,       &Ai[buf][r0][c0]);
    lds_async16(Aim + (size_t)(bm + r0 + 64) * K + k0 + c0,  &Ai[buf][r0 + 64][c0]);
    lds_async16(Wre + (size_t)(bn + r0) * K + k0 + c0,       &Wr[buf][r0][c0]);
    lds_async16(Wre + (size_t)(bn + r0 + 64) * K + k0 + c0,  &Wr[buf][r0 + 64][c0]);
    lds_async16(WimN + (size_t)(bn + r0) * K + k0 + c0,      &Wi[buf][r0][c0]);
    lds_async16(WimN + (size_t)(bn + r0 + 64) * K + k0 + c0, &Wi[buf][r0 + 64][c0]);
  };

  stage(0, 0);
  async_wait0();
  __syncthreads();

  const int nk = K >> 5;
  for (int kk = 0; kk < nk; ++kk) {
    const int buf = kk & 1;
    if (kk + 1 < nk) stage(buf ^ 1, (kk + 1) << 5);
    BF16Frag ar, ai;
    ar.h2[0] = *(const bf16x8*)&Ar[buf][aRow][aCol];
    ar.h2[1] = *(const bf16x8*)&Ar[buf][aRow][aCol + 16];
    ai.h2[0] = *(const bf16x8*)&Ai[buf][aRow][aCol];
    ai.h2[1] = *(const bf16x8*)&Ai[buf][aRow][aCol + 16];
#pragma unroll
    for (int j = 0; j < 8; ++j) {
      BF16Frag wr, wi;
      wr.h2[0] = *(const bf16x8*)&Wr[buf][j * 16 + l15][wCol];
      wr.h2[1] = *(const bf16x8*)&Wr[buf][j * 16 + l15][wCol + 16];
      wi.h2[0] = *(const bf16x8*)&Wi[buf][j * 16 + l15][wCol];
      wi.h2[1] = *(const bf16x8*)&Wi[buf][j * 16 + l15][wCol + 16];
      acc[j] = __builtin_amdgcn_wmma_f32_16x16x32_bf16(
          false, ar.v, false, wr.v, (short)0, acc[j], false, false);
      acc[j] = __builtin_amdgcn_wmma_f32_16x16x32_bf16(
          false, ai.v, false, wi.v, (short)0, acc[j], false, false);
    }
    if (kk + 1 < nk) {
      async_wait0();
      __syncthreads();
    }
  }

  const int mBase = bm + wave * 16 + hl * 8;
#pragma unroll
  for (int j = 0; j < 8; ++j) {
    const int n = bn + j * 16 + l15;
#pragma unroll
    for (int r = 0; r < 8; ++r)
      outp[(size_t)(mBase + r) * N + n] = acc[j][r];
  }
}

// ---------------------------------------------------------------- chunked complex scan
// state[t] = lam*state[t-1] + Bu[t], in-place on (sre,sim) [B,S,U]
__global__ void scan_phase1_k(float* __restrict__ sre, float* __restrict__ sim,
                              const float* __restrict__ nu, const float* __restrict__ th,
                              float* __restrict__ carry) {
  int tid = blockIdx.x * blockDim.x + threadIdx.x;
  if (tid >= BB * HH * NCK) return;
  int u = tid % HH;
  int c = (tid / HH) % NCK;
  int b = tid / (HH * NCK);
  float r = expf(-expf(nu[u]));
  float ph = expf(th[u]);
  float lre = r * cosf(ph), lim = r * sinf(ph);
  float zre = 0.f, zim = 0.f;
  size_t base = ((size_t)b * SS + (size_t)c * CHL) * HH + u;
  for (int t = 0; t < CHL; ++t) {
    size_t o = base + (size_t)t * HH;
    float nre = lre * zre - lim * zim + sre[o];
    float nim = lre * zim + lim * zre + sim[o];
    sre[o] = nre; sim[o] = nim;
    zre = nre; zim = nim;
  }
  size_t ci = (((size_t)b * HH + u) * NCK + c) * 2;
  carry[ci] = zre; carry[ci + 1] = zim;
}

__global__ void scan_phase2_k(const float* __restrict__ carry, float* __restrict__ zin,
                              const float* __restrict__ nu, const float* __restrict__ th) {
  int tid = blockIdx.x * blockDim.x + threadIdx.x;
  if (tid >= BB * HH) return;
  int u = tid % HH;
  float en = expf(nu[u]);
  float ph = expf(th[u]);
  float mag = expf(-(float)CHL * en);          // |lam|^CHL
  float ang = (float)CHL * ph;
  float Arep = mag * cosf(ang), Aimp = mag * sinf(ang);   // lam^CHL
  float zre = 0.f, zim = 0.f;
  size_t base = (size_t)tid * NCK * 2;
  for (int c = 0; c < NCK; ++c) {
    zin[base + 2 * c] = zre; zin[base + 2 * c + 1] = zim;
    float cre = carry[base + 2 * c], cim = carry[base + 2 * c + 1];
    float nre = Arep * zre - Aimp * zim + cre;
    float nim = Arep * zim + Aimp * zre + cim;
    zre = nre; zim = nim;
  }
}

__global__ void scan_phase3_k(float* __restrict__ sre, float* __restrict__ sim,
                              const float* __restrict__ zin,
                              const float* __restrict__ nu, const float* __restrict__ th) {
  int tid = blockIdx.x * blockDim.x + threadIdx.x;
  if (tid >= BB * HH * NCK) return;
  int u = tid % HH;
  int c = (tid / HH) % NCK;
  int b = tid / (HH * NCK);
  size_t zi = (((size_t)b * HH + u) * NCK + c) * 2;
  float zre = zin[zi], zim = zin[zi + 1];
  if (zre == 0.f && zim == 0.f && c == 0) return;
  float r = expf(-expf(nu[u]));
  float ph = expf(th[u]);
  float lre = r * cosf(ph), lim = r * sinf(ph);
  float pre = lre, pim = lim;                   // lam^(t_local+1)
  size_t base = ((size_t)b * SS + (size_t)c * CHL) * HH + u;
  for (int t = 0; t < CHL; ++t) {
    size_t o = base + (size_t)t * HH;
    sre[o] += pre * zre - pim * zim;
    sim[o] += pre * zim + pim * zre;
    float npre = pre * lre - pim * lim;
    float npim = pre * lim + pim * lre;
    pre = npre; pim = npim;
  }
}

// ---------------------------------------------------------------- fused residual + RMS-LN
// x = h + add (+ D*h);  h = x / sqrt(mean(x^2)+eps) + bias; emits bf16 copy; optional
// extra f32 sink (final model output).
__global__ __launch_bounds__(256) void ln_k(float* __restrict__ h,
                                            const float* __restrict__ add,
                                            const float* __restrict__ dvec,
                                            const float* __restrict__ bias,
                                            __bf16* __restrict__ hbf,
                                            float* __restrict__ extra_out) {
  __shared__ float red[256];
  const size_t row = blockIdx.x;
  float* hr = h + row * HH;
  const float* ar = add + row * HH;
  float v[4];
  float ss = 0.f;
#pragma unroll
  for (int i = 0; i < 4; ++i) {
    int idx = threadIdx.x + i * 256;
    float x = hr[idx] + ar[idx];
    if (dvec) x += dvec[idx] * hr[idx];
    v[i] = x;
    ss += x * x;
  }
  red[threadIdx.x] = ss;
  __syncthreads();
  for (int s = 128; s > 0; s >>= 1) {
    if (threadIdx.x < (unsigned)s) red[threadIdx.x] += red[threadIdx.x + s];
    __syncthreads();
  }
  float inv = rsqrtf(red[0] * (1.0f / HH) + 1e-12f);
#pragma unroll
  for (int i = 0; i < 4; ++i) {
    int idx = threadIdx.x + i * 256;
    float y = v[i] * inv + bias[idx];
    hr[idx] = y;
    hbf[row * HH + idx] = (__bf16)y;
    if (extra_out) extra_out[row * HH + idx] = y;
  }
}

// ---------------------------------------------------------------- host orchestration
extern "C" void kernel_launch(void* const* d_in, const int* in_sizes, int n_in,
                              void* d_out, int out_size, void* d_ws, size_t ws_size,
                              hipStream_t stream) {
  const int*   ids      = (const int*)d_in[0];
  const float* word_emb = (const float*)d_in[1];
  const float* pos_emb  = (const float*)d_in[2];
  const float* nu_log   = (const float*)d_in[3];
  const float* theta_log= (const float*)d_in[4];
  const float* B_re     = (const float*)d_in[5];
  const float* B_im     = (const float*)d_in[6];
  const float* C_re     = (const float*)d_in[7];
  const float* C_im     = (const float*)d_in[8];
  const float* Dv       = (const float*)d_in[9];
  const float* rnn_bias = (const float*)d_in[10];
  const float* w1       = (const float*)d_in[11];
  const float* b1       = (const float*)d_in[12];
  const float* w2       = (const float*)d_in[13];
  const float* b2       = (const float*)d_in[14];
  const float* ffn_bias = (const float*)d_in[15];

  char* ws = (char*)d_ws;
  size_t off = 0;
  auto carve = [&](size_t bytes) {
    size_t r = off;
    off += (bytes + 255) & ~(size_t)255;
    return r;
  };
  const size_t HxH = (size_t)HH * HH, FxH = (size_t)FF * HH;
  const size_t MT = (size_t)BB * SS;              // 8192 rows
  __bf16* wBre  = (__bf16*)(ws + carve(LL * HxH * 2));
  __bf16* wBim  = (__bf16*)(ws + carve(LL * HxH * 2));
  __bf16* wCre  = (__bf16*)(ws + carve(LL * HxH * 2));
  __bf16* wCimN = (__bf16*)(ws + carve(LL * HxH * 2));  // negated C_im
  __bf16* ww1   = (__bf16*)(ws + carve(LL * FxH * 2));
  __bf16* ww2   = (__bf16*)(ws + carve(LL * FxH * 2));
  float*  h     = (float*) (ws + carve(MT * HH * 4));
  __bf16* h_bf  = (__bf16*)(ws + carve(MT * HH * 2));
  float*  s_re  = (float*) (ws + carve(MT * HH * 4));
  float*  s_im  = (float*) (ws + carve(MT * HH * 4));
  __bf16* sre_bf= (__bf16*)(ws + carve(MT * HH * 2));
  __bf16* sim_bf= (__bf16*)(ws + carve(MT * HH * 2));
  float*  ybuf  = (float*) (ws + carve(MT * HH * 4));   // rnn proj / ffn out (reused)
  __bf16* x1_bf = (__bf16*)(ws + carve(MT * FF * 2));
  float*  gam   = (float*) (ws + carve((size_t)LL * HH * 4));
  float*  carry = (float*) (ws + carve((size_t)BB * HH * NCK * 2 * 4));
  float*  zin   = (float*) (ws + carve((size_t)BB * HH * NCK * 2 * 4));

  // one-time weight conversions (deterministic every call)
  cvt_f32_bf16_k<<<8192, 256, 0, stream>>>(B_re, wBre, LL * HxH, 1.0f);
  cvt_f32_bf16_k<<<8192, 256, 0, stream>>>(B_im, wBim, LL * HxH, 1.0f);
  cvt_f32_bf16_k<<<8192, 256, 0, stream>>>(C_re, wCre, LL * HxH, 1.0f);
  cvt_f32_bf16_k<<<8192, 256, 0, stream>>>(C_im, wCimN, LL * HxH, -1.0f);
  cvt_f32_bf16_k<<<8192, 256, 0, stream>>>(w1, ww1, LL * FxH, 1.0f);
  cvt_f32_bf16_k<<<8192, 256, 0, stream>>>(w2, ww2, LL * FxH, 1.0f);
  gamma_k<<<(LL * HH + 255) / 256, 256, 0, stream>>>(nu_log, gam, LL * HH);
  embed_k<<<(unsigned)((MT * HH + 255) / 256), 256, 0, stream>>>(ids, word_emb, pos_emb, h, h_bf);

  const dim3 gH((unsigned)(MT / 128), HH / 128);   // 64 x 8
  const dim3 gF((unsigned)(MT / 128), FF / 128);   // 64 x 32
  const int scanBlocks = (BB * HH * NCK + 255) / 256;

  for (int l = 0; l < LL; ++l) {
    const float* nu_l = nu_log + (size_t)l * HH;
    const float* th_l = theta_log + (size_t)l * HH;
    // Bu = (h @ B^T) * gamma  (re and im)
    gemm_bf16_k<0><<<gH, 256, 0, stream>>>(h_bf, wBre + l * HxH, gam + (size_t)l * HH, s_re, HH, HH);
    gemm_bf16_k<0><<<gH, 256, 0, stream>>>(h_bf, wBim + l * HxH, gam + (size_t)l * HH, s_im, HH, HH);
    // diagonal complex linear recurrence (chunked parallel scan)
    scan_phase1_k<<<scanBlocks, 256, 0, stream>>>(s_re, s_im, nu_l, th_l, carry);
    scan_phase2_k<<<(BB * HH + 255) / 256, 256, 0, stream>>>(carry, zin, nu_l, th_l);
    scan_phase3_k<<<scanBlocks, 256, 0, stream>>>(s_re, s_im, zin, nu_l, th_l);
    // bf16 copies of states for the WMMA projection
    cvt_f32_bf16_k<<<8192, 256, 0, stream>>>(s_re, sre_bf, MT * HH, 1.0f);
    cvt_f32_bf16_k<<<8192, 256, 0, stream>>>(s_im, sim_bf, MT * HH, 1.0f);
    // y = Re(s) @ C_re^T - Im(s) @ C_im^T
    gemm_pair_k<<<gH, 256, 0, stream>>>(sre_bf, sim_bf, wCre + l * HxH, wCimN + l * HxH, ybuf, HH, HH);
    // h = LN(h + y + D*h) + bias
    ln_k<<<(unsigned)MT, 256, 0, stream>>>(h, ybuf, Dv + (size_t)l * HH,
                                           rnn_bias + (size_t)l * HH, h_bf, nullptr);
    // FFN: x1 = gelu(h @ w1^T + b1);  ffn = x1 @ w2^T + b2
    gemm_bf16_k<1><<<gF, 256, 0, stream>>>(h_bf, ww1 + l * FxH, b1 + (size_t)l * FF, x1_bf, FF, HH);
    gemm_bf16_k<2><<<gH, 256, 0, stream>>>(x1_bf, ww2 + l * FxH, b2 + (size_t)l * HH, ybuf, HH, FF);
    // h = LN(h + ffn) + bias   (last layer also streams result into d_out)
    ln_k<<<(unsigned)MT, 256, 0, stream>>>(h, ybuf, nullptr, ffn_bias + (size_t)l * HH, h_bf,
                                           (l == LL - 1) ? (float*)d_out : nullptr);
  }
}
